// AttnPathDecoder_63050119905368
// MI455X (gfx1250) — matrix operand order repaired and checked by reading, not compiled, not run on hardware
//
#include <hip/hip_runtime.h>

#define DEVFN __device__ __forceinline__

typedef __bf16 bf16_t;
typedef __attribute__((ext_vector_type(16))) __bf16 v16bf;
typedef __attribute__((ext_vector_type(8)))  __bf16 v8bf;
typedef __attribute__((ext_vector_type(8)))  float  v8f;
typedef __attribute__((ext_vector_type(4)))  int    v4i;

// ---- problem constants (fixed by the reference) ----
constexpr int BB = 2;      // batch
constexpr int TT = 2048;   // Td == Te
constexpr int DD = 512;    // model dim
constexpr int FF = 2048;   // ffn dim
constexpr int HH = 8;      // heads
constexpr int DH = 64;     // head dim
constexpr int LL = 2;      // layers
constexpr int MM = BB * TT; // 4096 rows of the activation matrix

// ---------------------------------------------------------------------------
// CDNA5 async global->LDS copy (ASYNCcnt path), with safe fallback
// Builtin signature (from compiler diagnostic): (v4i as1*, v4i as3*, Ii, Ii)
// ---------------------------------------------------------------------------
#if __has_builtin(__builtin_amdgcn_global_load_async_to_lds_b128)
#define HAVE_ASYNC 1
#endif

#if defined(HAVE_ASYNC)
#if __has_builtin(__builtin_amdgcn_s_wait_asynccnt)
#define WAIT_ASYNC(n) __builtin_amdgcn_s_wait_asynccnt(n)
#else
#define WAIT_ASYNC(n) asm volatile("s_wait_asynccnt %0" ::"i"(n) : "memory")
#endif
#else
#define WAIT_ASYNC(n) ((void)0)
#endif

DEVFN void async_copy16(const bf16_t* g, bf16_t* l) {
#if defined(HAVE_ASYNC)
  __builtin_amdgcn_global_load_async_to_lds_b128(
      (__attribute__((address_space(1))) v4i*)(g),
      (__attribute__((address_space(3))) v4i*)(l), 0, 0);
#else
  *(v8bf*)l = *(const v8bf*)g;   // synchronous fallback
#endif
}

// ---------------------------------------------------------------------------
// WMMA helpers (CDNA5 16x16x32 bf16, f32 accumulate)
// ---------------------------------------------------------------------------
DEVFN v8f wmma_bf16(v16bf a, v16bf b, v8f c) {
  return __builtin_amdgcn_wmma_f32_16x16x32_bf16(
      /*neg_a=*/false, a, /*neg_b=*/false, b,
      /*c_mod=*/(short)0, c, /*reuse_a=*/false, /*reuse_b=*/false);
}

// A-fragment, 16x32 bf16 (M x K), row-major source with leading dim `lda`.
// Contiguous-k layout used here: lane m = lane&15; k = e + 16*(lane>=16)
DEVFN v16bf load_a_frag(const bf16_t* __restrict__ src, int lda, int lane) {
  int m  = lane & 15;
  int k0 = (lane >> 4) << 4;          // 0 or 16
  const bf16_t* p = src + (size_t)m * lda + k0;
  v8bf lo = *(const v8bf*)(p);        // k0..k0+7
  v8bf hi = *(const v8bf*)(p + 8);    // k0+8..k0+15
  return __builtin_shufflevector(lo, hi, 0,1,2,3,4,5,6,7,8,9,10,11,12,13,14,15);
}

// B-fragment, 32x16 bf16 (K x N). Source stored TRANSPOSED: srcT[n][k].
// Contiguous-k layout used here: lane n = lane&15; k = e + 16*(lane>=16)
DEVFN v16bf load_b_frag(const bf16_t* __restrict__ srcT, int ldb, int lane) {
  int n  = lane & 15;
  int k0 = (lane >> 4) << 4;          // 0 or 16
  return *(const v16bf*)(srcT + (size_t)n * ldb + k0);
}

// LDS variants with padded leading dim 40 (80B = 20 banks: conflict-free rows)
constexpr int LDSP = 40;
DEVFN v16bf frag_a_lds(const bf16_t* rowbase, int lane) {
  int m = lane & 15, k0 = (lane >> 4) << 4;
  const bf16_t* p = rowbase + m * LDSP + k0;
  v8bf lo = *(const v8bf*)(p);
  v8bf hi = *(const v8bf*)(p + 8);
  return __builtin_shufflevector(lo, hi, 0,1,2,3,4,5,6,7,8,9,10,11,12,13,14,15);
}
DEVFN v16bf frag_b_lds(const bf16_t* colbase, int lane) {
  int n = lane & 15, k0 = (lane >> 4) << 4;
  const bf16_t* p = colbase + n * LDSP + k0;
  v8bf lo = *(const v8bf*)(p);
  v8bf hi = *(const v8bf*)(p + 8);
  return __builtin_shufflevector(lo, hi, 0,1,2,3,4,5,6,7,8,9,10,11,12,13,14,15);
}

// ---------------------------------------------------------------------------
// small utility kernels
// ---------------------------------------------------------------------------
__global__ void k_zero_f32(float* __restrict__ p, int n) {
  int i = blockIdx.x * 256 + threadIdx.x;
  if (i < n) p[i] = 0.0f;
}
__global__ void k_copy_f32(const float* __restrict__ a, float* __restrict__ b, int n) {
  int i = blockIdx.x * 256 + threadIdx.x;
  if (i < n) b[i] = a[i];
}
__global__ void k_f32_to_bf16(const float* __restrict__ a, bf16_t* __restrict__ b, int n) {
  int i = blockIdx.x * 256 + threadIdx.x;
  if (i < n) b[i] = (bf16_t)a[i];
}
// W[din][dout] fp32 -> WT[dout][din] bf16  (coalesced writes along din)
__global__ void k_wconv(const float* __restrict__ w, bf16_t* __restrict__ wt,
                        int din, int dout) {
  int i = blockIdx.x * 32 + threadIdx.x;  // din index
  int o = blockIdx.y * 8  + threadIdx.y;  // dout index
  if (i < din && o < dout)
    wt[(size_t)o * din + i] = (bf16_t)w[(size_t)i * dout + o];
}

// ---------------------------------------------------------------------------
// LayerNorm: one block per row (D=512, 256 threads x 2 elems)
// ---------------------------------------------------------------------------
template<bool BF16OUT>
__global__ __launch_bounds__(256)
void k_layernorm(const float* __restrict__ x, const float* __restrict__ g,
                 const float* __restrict__ beta,
                 bf16_t* __restrict__ ob, float* __restrict__ of) {
  __shared__ float s1[256], s2[256];
  int row = blockIdx.x, tid = threadIdx.x;
  const float* xr = x + (size_t)row * DD;
  float a0 = xr[tid], a1 = xr[tid + 256];
  s1[tid] = a0 + a1;
  s2[tid] = a0 * a0 + a1 * a1;
  __syncthreads();
  for (int s = 128; s > 0; s >>= 1) {
    if (tid < s) { s1[tid] += s1[tid + s]; s2[tid] += s2[tid + s]; }
    __syncthreads();
  }
  float mean = s1[0] * (1.0f / DD);
  float var  = s2[0] * (1.0f / DD) - mean * mean;
  float rstd = rsqrtf(var + 1e-6f);
  float y0 = g[tid]       * (a0 - mean) * rstd + beta[tid];
  float y1 = g[tid + 256] * (a1 - mean) * rstd + beta[tid + 256];
  if (BF16OUT) {
    ob[(size_t)row * DD + tid]       = (bf16_t)y0;
    ob[(size_t)row * DD + tid + 256] = (bf16_t)y1;
  } else {
    of[(size_t)row * DD + tid]       = y0;
    of[(size_t)row * DD + tid + 256] = y1;
  }
}

// ---------------------------------------------------------------------------
// WMMA GEMM, LDS-staged + async double buffer.
//   C[M=4096][N] = A[M][K] (bf16 row-major) * WT[N][K]^T
// Block tile 128x64, 8 waves as 4(m)x2(n), each wave 2x2 WMMA fragments.
// Per k-step each thread issues 3 async B128 global->LDS copies; compute
// overlaps the next stage's DMA (steady-state wait: asynccnt <= 3).
// MODE 0: bf16 out, [B][H][T][dh] (Q/K, scaled)   MODE 1: [B][H][dh][T] (V^T)
// MODE 2: bf16 ReLU row-major (FFN mid)           MODE 3: fp32 resid += acc
// ---------------------------------------------------------------------------
template<int MODE, int N>
DEVFN void store_tile(v8f acc, int mbase, int nbase, int lane, float scale,
                      float* __restrict__ resid, bf16_t* __restrict__ outb) {
  int n = nbase + (lane & 15);
#pragma unroll
  for (int r = 0; r < 8; ++r) {
    int m = mbase + r + ((lane >> 4) << 3);
    float val = acc[r] * scale;
    if (MODE == 3) {
      resid[(size_t)m * N + n] += val;
    } else if (MODE == 2) {
      outb[(size_t)m * N + n] = (bf16_t)(val > 0.0f ? val : 0.0f);
    } else {
      int b = m >> 11, t = m & (TT - 1);
      int h = n >> 6,  d = n & (DH - 1);
      size_t idx = (MODE == 0)
          ? ((((size_t)b * HH + h) * TT + t) * DH + d)
          : ((((size_t)b * HH + h) * DH + d) * TT + t);
      outb[idx] = (bf16_t)val;
    }
  }
}

template<int MODE, int N, int K>
__global__ __launch_bounds__(256)
void k_gemm(const bf16_t* __restrict__ A, const bf16_t* __restrict__ BT,
            float scale, float* __restrict__ resid, bf16_t* __restrict__ outb) {
  __shared__ bf16_t sA[2][128][LDSP];   // 128 rows x 32 k (padded)
  __shared__ bf16_t sB[2][64][LDSP];    // 64 cols  x 32 k (padded)

  int wave = threadIdx.x >> 5, lane = threadIdx.x & 31;
  int wm = wave >> 1, wn = wave & 1;           // 4 x 2 wave grid
  int mblock = blockIdx.y * 128;
  int nblock = blockIdx.x * 64;

  auto issue = [&](int stage, int kk) {
    int buf = stage & 1;
    // A tile: 128 rows x 64B = 512 x 16B units, 2 per thread
#pragma unroll
    for (int j = 0; j < 2; ++j) {
      int u = threadIdx.x + j * 256;
      int row = u >> 2, seg = u & 3;
      async_copy16(A + (size_t)(mblock + row) * K + kk + seg * 8,
                   &sA[buf][row][seg * 8]);
    }
    // B tile: 64 rows x 64B = 256 x 16B units, 1 per thread
    {
      int row = threadIdx.x >> 2, seg = threadIdx.x & 3;
      async_copy16(BT + (size_t)(nblock + row) * K + kk + seg * 8,
                   &sB[buf][row][seg * 8]);
    }
  };

  v8f acc00 = {0.f,0.f,0.f,0.f,0.f,0.f,0.f,0.f};
  v8f acc01 = acc00, acc10 = acc00, acc11 = acc00;

  constexpr int S = K / 32;   // 16 or 64 stages
  issue(0, 0);
  issue(1, 32);
  for (int i = 0; i < S; ++i) {
    if (i + 1 < S) { WAIT_ASYNC(3); } else { WAIT_ASYNC(0); }
    __syncthreads();
    int buf = i & 1;
    v16bf a0 = frag_a_lds(&sA[buf][wm * 32 +  0][0], lane);
    v16bf a1 = frag_a_lds(&sA[buf][wm * 32 + 16][0], lane);
    v16bf b0 = frag_b_lds(&sB[buf][wn * 32 +  0][0], lane);
    v16bf b1 = frag_b_lds(&sB[buf][wn * 32 + 16][0], lane);
    acc00 = wmma_bf16(a0, b0, acc00);
    acc01 = wmma_bf16(a0, b1, acc01);
    acc10 = wmma_bf16(a1, b0, acc10);
    acc11 = wmma_bf16(a1, b1, acc11);
    __syncthreads();          // protect buf before stage i+2 overwrites it
    if (i + 2 < S) issue(i + 2, (i + 2) * 32);
  }

  int mb = mblock + wm * 32, nb = nblock + wn * 32;
  store_tile<MODE, N>(acc00, mb,      nb,      lane, scale, resid, outb);
  store_tile<MODE, N>(acc01, mb,      nb + 16, lane, scale, resid, outb);
  store_tile<MODE, N>(acc10, mb + 16, nb,      lane, scale, resid, outb);
  store_tile<MODE, N>(acc11, mb + 16, nb + 16, lane, scale, resid, outb);
}

// ---------------------------------------------------------------------------
// Attention: one block per (b, h, 16-row q tile). Exact softmax using full
// score rows in LDS (16 x 2048 fp32 + 16 x 2048 bf16 probs = 192KB of the
// 320KB WGP LDS). Optionally accumulates head-averaged weights into `avg`.
//   q : [B][H][T][dh] bf16 (pre-scaled)   kk: [B][H][T][dh]   vT: [B][H][dh][T]
//   out: [B*T][D] bf16 row-major (col = h*64 + d)
// ---------------------------------------------------------------------------
constexpr size_t ATTN_SMEM = (size_t)16 * TT * 4 + (size_t)16 * TT * 2;

__global__ __launch_bounds__(256)
void k_attn(const bf16_t* __restrict__ q, const bf16_t* __restrict__ kk,
            const bf16_t* __restrict__ vT, bf16_t* __restrict__ out,
            float* __restrict__ avg, int causal) {
  extern __shared__ char smem[];
  float*  scores = (float*)smem;                          // [16][TT]
  bf16_t* probs  = (bf16_t*)(smem + (size_t)16 * TT * 4); // [16][TT]
  __shared__ float red[16][17];

  int b = blockIdx.z, h = blockIdx.y, q0 = blockIdx.x << 4;
  int wave = threadIdx.x >> 5, lane = threadIdx.x & 31;

  const bf16_t* qbase = q + ((((size_t)b * HH + h) * TT) + q0) * DH;
  v16bf qf0 = load_a_frag(qbase + 0,  DH, lane);
  v16bf qf1 = load_a_frag(qbase + 32, DH, lane);

  // ---- phase 1: S = Q K^T ----
  const bf16_t* kbase = kk + (((size_t)b * HH + h) * TT) * DH;
  for (int tt = wave; tt < (TT >> 4); tt += 8) {
    int t0 = tt << 4;
    v16bf kb0 = load_b_frag(kbase + (size_t)t0 * DH + 0,  DH, lane);
    v16bf kb1 = load_b_frag(kbase + (size_t)t0 * DH + 32, DH, lane);
    v8f s = {0.f,0.f,0.f,0.f,0.f,0.f,0.f,0.f};
    s = wmma_bf16(qf0, kb0, s);
    s = wmma_bf16(qf1, kb1, s);
    int tg = t0 + (lane & 15);
#pragma unroll
    for (int r = 0; r < 8; ++r) {
      int m = r + ((lane >> 4) << 3);
      float val = s[r];
      if (causal && tg > q0 + m) val = -1e9f;
      scores[(size_t)m * TT + tg] = val;
    }
  }
  __syncthreads();

  // ---- phase 2: exact softmax, 16 threads per row ----
  int row = threadIdx.x >> 4, seg = threadIdx.x & 15;
  int c0 = seg * (TT >> 4);
  float mx = -1e30f;
  for (int c = c0; c < c0 + (TT >> 4); ++c)
    mx = fmaxf(mx, scores[(size_t)row * TT + c]);
  red[row][seg] = mx;
  __syncthreads();
  if (seg == 0) {
    float m2 = red[row][0];
    for (int i = 1; i < 16; ++i) m2 = fmaxf(m2, red[row][i]);
    red[row][16] = m2;
  }
  __syncthreads();
  float rowmax = red[row][16];
  float sm = 0.0f;
  for (int c = c0; c < c0 + (TT >> 4); ++c)
    sm += __expf(scores[(size_t)row * TT + c] - rowmax);
  __syncthreads();
  red[row][seg] = sm;
  __syncthreads();
  if (seg == 0) {
    float s2 = 0.0f;
    for (int i = 0; i < 16; ++i) s2 += red[row][i];
    red[row][16] = s2;
  }
  __syncthreads();
  float inv = 1.0f / red[row][16];
  const float cf = 1.0f / (float)(HH * LL);
  for (int c = c0; c < c0 + (TT >> 4); ++c) {
    float p = __expf(scores[(size_t)row * TT + c] - rowmax) * inv;
    probs[(size_t)row * TT + c] = (bf16_t)p;
    if (avg) atomicAdd(&avg[((size_t)b * TT + q0 + row) * TT + c], p * cf);
  }
  __syncthreads();

  // ---- phase 3: O = P V ----
  if (wave < 4) {
    int d0 = wave << 4;
    const bf16_t* vbase = vT + (((size_t)b * HH + h) * DH) * TT;  // [64][TT]
    v8f o = {0.f,0.f,0.f,0.f,0.f,0.f,0.f,0.f};
    for (int kc = 0; kc < TT; kc += 32) {
      v16bf pa = load_a_frag(probs + kc, TT, lane);
      v16bf vb = load_b_frag(vbase + (size_t)d0 * TT + kc, TT, lane);
      o = wmma_bf16(pa, vb, o);
    }
    int n = lane & 15;
#pragma unroll
    for (int r = 0; r < 8; ++r) {
      int m = r + ((lane >> 4) << 3);
      out[((size_t)b * TT + q0 + m) * DD + h * DH + d0 + n] = (bf16_t)o[r];
    }
  }
}

// ---------------------------------------------------------------------------
// host-side orchestration
// ---------------------------------------------------------------------------
static inline void* ws_take(char* base, size_t& off, size_t bytes) {
  off = (off + 255) & ~(size_t)255;
  void* p = base + off;
  off += bytes;
  return p;
}

extern "C" void kernel_launch(void* const* d_in, const int* in_sizes, int n_in,
                              void* d_out, int out_size, void* d_ws, size_t ws_size,
                              hipStream_t stream) {
  (void)in_sizes; (void)n_in; (void)out_size; (void)ws_size;

  const float* x_in  = (const float*)d_in[0];
  const float* enc_a = (const float*)d_in[1];
  const float* enc_c = (const float*)d_in[2];
  const float* sa_wq = (const float*)d_in[3];
  const float* sa_wk = (const float*)d_in[4];
  const float* sa_wv = (const float*)d_in[5];
  const float* sa_wo = (const float*)d_in[6];
  const float* ed_wq = (const float*)d_in[7];
  const float* ed_wk = (const float*)d_in[8];
  const float* ed_wv = (const float*)d_in[9];
  const float* ed_wo = (const float*)d_in[10];
  const float* f_w1  = (const float*)d_in[11];
  const float* f_w2  = (const float*)d_in[12];
  const float* ln1_g = (const float*)d_in[13];
  const float* ln1_b = (const float*)d_in[14];
  const float* ln2_g = (const float*)d_in[15];
  const float* ln2_b = (const float*)d_in[16];
  const float* ln3_g = (const float*)d_in[17];
  const float* ln3_b = (const float*)d_in[18];
  const float* out_g = (const float*)d_in[19];
  const float* out_b = (const float*)d_in[20];

  float* out_x   = (float*)d_out;                 // [B][T][D]
  float* out_avg = out_x + (size_t)MM * DD;       // [B][T][T]

  char* ws = (char*)d_ws;
  size_t off = 0;
  float*  xbuf  = (float*) ws_take(ws, off, (size_t)MM * DD * 4);
  bf16_t* hbuf  = (bf16_t*)ws_take(ws, off, (size_t)MM * DD * 2);
  bf16_t* aobuf = (bf16_t*)ws_take(ws, off, (size_t)MM * DD * 2);
  bf16_t* qb    = (bf16_t*)ws_take(ws, off, (size_t)MM * DD * 2);
  bf16_t* kb    = (bf16_t*)ws_take(ws, off, (size_t)MM * DD * 2);
  bf16_t* vb    = (bf16_t*)ws_take(ws, off, (size_t)MM * DD * 2);
  bf16_t* mid   = (bf16_t*)ws_take(ws, off, (size_t)MM * FF * 2);
  bf16_t* ea_bf = (bf16_t*)ws_take(ws, off, (size_t)MM * DD * 2);
  bf16_t* ec_bf = (bf16_t*)ws_take(ws, off, (size_t)MM * DD * 2);
  bf16_t* wT_saq = (bf16_t*)ws_take(ws, off, (size_t)LL * DD * DD * 2);
  bf16_t* wT_sak = (bf16_t*)ws_take(ws, off, (size_t)LL * DD * DD * 2);
  bf16_t* wT_sav = (bf16_t*)ws_take(ws, off, (size_t)LL * DD * DD * 2);
  bf16_t* wT_sao = (bf16_t*)ws_take(ws, off, (size_t)LL * DD * DD * 2);
  bf16_t* wT_edq = (bf16_t*)ws_take(ws, off, (size_t)LL * DD * DD * 2);
  bf16_t* wT_edk = (bf16_t*)ws_take(ws, off, (size_t)LL * DD * DD * 2);
  bf16_t* wT_edv = (bf16_t*)ws_take(ws, off, (size_t)LL * DD * DD * 2);
  bf16_t* wT_edo = (bf16_t*)ws_take(ws, off, (size_t)LL * DD * DD * 2);
  bf16_t* wT_f1  = (bf16_t*)ws_take(ws, off, (size_t)LL * FF * DD * 2); // [F][D]
  bf16_t* wT_f2  = (bf16_t*)ws_take(ws, off, (size_t)LL * DD * FF * 2); // [D][F]

  const int nElems = MM * DD;
  k_copy_f32<<<(nElems + 255) / 256, 256, 0, stream>>>(x_in, xbuf, nElems);
  k_zero_f32<<<(BB * TT * TT + 255) / 256, 256, 0, stream>>>(out_avg, BB * TT * TT);
  k_f32_to_bf16<<<(nElems + 255) / 256, 256, 0, stream>>>(enc_a, ea_bf, nElems);
  k_f32_to_bf16<<<(nElems + 255) / 256, 256, 0, stream>>>(enc_c, ec_bf, nElems);

  dim3 wcb(32, 8);
  for (int i = 0; i < LL; ++i) {
    size_t wo = (size_t)i * DD * DD;
    k_wconv<<<dim3(DD / 32, DD / 8), wcb, 0, stream>>>(sa_wq + wo, wT_saq + wo, DD, DD);
    k_wconv<<<dim3(DD / 32, DD / 8), wcb, 0, stream>>>(sa_wk + wo, wT_sak + wo, DD, DD);
    k_wconv<<<dim3(DD / 32, DD / 8), wcb, 0, stream>>>(sa_wv + wo, wT_sav + wo, DD, DD);
    k_wconv<<<dim3(DD / 32, DD / 8), wcb, 0, stream>>>(sa_wo + wo, wT_sao + wo, DD, DD);
    k_wconv<<<dim3(DD / 32, DD / 8), wcb, 0, stream>>>(ed_wq + wo, wT_edq + wo, DD, DD);
    k_wconv<<<dim3(DD / 32, DD / 8), wcb, 0, stream>>>(ed_wk + wo, wT_edk + wo, DD, DD);
    k_wconv<<<dim3(DD / 32, DD / 8), wcb, 0, stream>>>(ed_wv + wo, wT_edv + wo, DD, DD);
    k_wconv<<<dim3(DD / 32, DD / 8), wcb, 0, stream>>>(ed_wo + wo, wT_edo + wo, DD, DD);
    size_t fo = (size_t)i * DD * FF;
    k_wconv<<<dim3(DD / 32, FF / 8), wcb, 0, stream>>>(f_w1 + fo, wT_f1 + fo, DD, FF);
    k_wconv<<<dim3(FF / 32, DD / 8), wcb, 0, stream>>>(f_w2 + fo, wT_f2 + fo, FF, DD);
  }

  const dim3 gDD(DD / 64, MM / 128);   // N=512 GEMMs (block tile 128x64)
  const dim3 gFF(FF / 64, MM / 128);   // N=2048 GEMM
  const dim3 gAttn(TT / 16, HH, BB);
  const float qscale = 0.125f;         // dh^-0.5, dh=64

  for (int i = 0; i < LL; ++i) {
    size_t wo = (size_t)i * DD * DD;
    size_t fo = (size_t)i * DD * FF;
    size_t lo = (size_t)i * DD;

    // ---- self-attention ----
    k_layernorm<true><<<MM, 256, 0, stream>>>(xbuf, ln1_g + lo, ln1_b + lo, hbuf, nullptr);
    k_gemm<0, DD, DD><<<gDD, 256, 0, stream>>>(hbuf, wT_saq + wo, qscale, nullptr, qb);
    k_gemm<0, DD, DD><<<gDD, 256, 0, stream>>>(hbuf, wT_sak + wo, 1.0f, nullptr, kb);
    k_gemm<1, DD, DD><<<gDD, 256, 0, stream>>>(hbuf, wT_sav + wo, 1.0f, nullptr, vb);
    k_attn<<<gAttn, 256, ATTN_SMEM, stream>>>(qb, kb, vb, aobuf, nullptr, 1);
    k_gemm<3, DD, DD><<<gDD, 256, 0, stream>>>(aobuf, wT_sao + wo, 1.0f, xbuf, nullptr);

    // ---- cross-attention (enc_a, then enc_c with weight accumulation) ----
    k_layernorm<true><<<MM, 256, 0, stream>>>(xbuf, ln2_g + lo, ln2_b + lo, hbuf, nullptr);
    k_gemm<0, DD, DD><<<gDD, 256, 0, stream>>>(hbuf, wT_edq + wo, qscale, nullptr, qb);

    k_gemm<0, DD, DD><<<gDD, 256, 0, stream>>>(ea_bf, wT_edk + wo, 1.0f, nullptr, kb);
    k_gemm<1, DD, DD><<<gDD, 256, 0, stream>>>(ea_bf, wT_edv + wo, 1.0f, nullptr, vb);
    k_attn<<<gAttn, 256, ATTN_SMEM, stream>>>(qb, kb, vb, aobuf, nullptr, 0);
    k_gemm<3, DD, DD><<<gDD, 256, 0, stream>>>(aobuf, wT_edo + wo, 1.0f, xbuf, nullptr);

    k_gemm<0, DD, DD><<<gDD, 256, 0, stream>>>(ec_bf, wT_edk + wo, 1.0f, nullptr, kb);
    k_gemm<1, DD, DD><<<gDD, 256, 0, stream>>>(ec_bf, wT_edv + wo, 1.0f, nullptr, vb);
    k_attn<<<gAttn, 256, ATTN_SMEM, stream>>>(qb, kb, vb, aobuf, out_avg, 0);
    k_gemm<3, DD, DD><<<gDD, 256, 0, stream>>>(aobuf, wT_edo + wo, 1.0f, xbuf, nullptr);

    // ---- FFN ----
    k_layernorm<true><<<MM, 256, 0, stream>>>(xbuf, ln3_g + lo, ln3_b + lo, hbuf, nullptr);
    k_gemm<2, FF, DD><<<gFF, 256, 0, stream>>>(hbuf, wT_f1 + fo, 1.0f, nullptr, mid);
    k_gemm<3, DD, FF><<<gDD, 256, 0, stream>>>(mid, wT_f2 + fo, 1.0f, xbuf, nullptr);
  }

  // final layernorm -> fp32 output
  k_layernorm<false><<<MM, 256, 0, stream>>>(xbuf, out_g, out_b, nullptr, out_x);
}
